// Model_39676907880909
// MI455X (gfx1250) — matrix operand order repaired and checked by reading, not compiled, not run on hardware
//
#include <hip/hip_runtime.h>
#include <hip/hip_bf16.h>

typedef __attribute__((ext_vector_type(16))) __bf16 v16bf;
typedef __attribute__((ext_vector_type(8)))  float  v8f;
typedef __attribute__((ext_vector_type(4)))  unsigned int v4u;
typedef __attribute__((ext_vector_type(8)))  int v8i;
typedef __attribute__((ext_vector_type(4)))  int v4i;

#define NHEAD 16
#define SEQ   2048
#define DH    128
#define BM    64          // queries per workgroup (4 waves x 16 rows)
#define BC    32          // keys per inner iteration
#define NKB   (SEQ / BC)  // 64 key blocks
#define KSTRIDE 132       // 128 + 4 f32 -> 528B rows (16B aligned, bank-conflict-free)
#define VPAD  40          // 32 + 8 bf16 -> 80B rows
#define PPAD  40
#define SCALE 0.08838834764831845f   // 1/sqrt(128)
#define KEEP_THRESH 15099494u        // 0.9 * 2^24
#define INV_KEEP (1.0f / 0.9f)

#if __has_builtin(__builtin_amdgcn_tensor_load_to_lds)
#define USE_TDM 1
#else
#define USE_TDM 0
#endif

__device__ __forceinline__ unsigned int mix_hash(unsigned int x) {
    x ^= x >> 16; x *= 0x7feb352du;
    x ^= x >> 15; x *= 0x846ca68bu;
    x ^= x >> 16;
    return x;
}

#if USE_TDM
// DMA one 32x128 f32 K tile (row-major) global->LDS via the Tensor Data Mover.
// D# per CDNA5 ISA 8.3/8.4: count=1, type=2(image), data_size=4B,
// tensor_dim0=128, dim0_stride=128, tile 128x32,
// pad_enable: +4 DWORDs every 128 DWORDs -> LDS rows of 132 f32 (528B).
__device__ __forceinline__ void tdm_load_k_tile(const float* gsrc, unsigned lds_off) {
    unsigned long long ga = (unsigned long long)(size_t)gsrc;
    v4u g0;
    g0.x = 1u;                                                  // count=1 (valid)
    g0.y = lds_off;                                             // lds_addr
    g0.z = (unsigned)(ga & 0xffffffffu);                        // global_addr lo
    g0.w = (unsigned)((ga >> 32) & 0x01ffffffu) | (2u << 30);   // global_addr hi | type=2
    v8i g1;
    g1[0] = (2 << 16) | (1 << 20) | (6 << 22) | (3 << 25);      // data_size=4B, pad 4dw/128dw
    g1[1] = (DH & 0xffff) << 16;                                // tensor_dim0 = 128 (lo16)
    g1[2] = (SEQ & 0xffff) << 16;                               // dim0 hi=0 | tensor_dim1 lo16
    g1[3] = (DH & 0xffff) << 16;                                // dim1 hi=0 | tile_dim0 = 128
    g1[4] = BC;                                                 // tile_dim1 = 32, tile_dim2 = 0
    g1[5] = DH;                                                 // tensor_dim0_stride lo32 = 128
    g1[6] = 0;
    g1[7] = 0;
    v4i g2 = {};
    v4i g3 = {};
#if __has_include(<hip/amd_detail/amd_gfx1250_TDM.h>)
    v8i g4 = {};
    __builtin_amdgcn_tensor_load_to_lds(g0, g1, g2, g3, g4, 0); // clang-23 6-arg form
#else
    __builtin_amdgcn_tensor_load_to_lds(g0, g1, g2, g3, 0);     // ROCm 7.2 5-arg form
#endif
}
#endif

__global__ __launch_bounds__(128)
void fattn_bf16_wmma(const float* __restrict__ q,
                     const float* __restrict__ k,
                     const float* __restrict__ v,
                     const float* __restrict__ mask,
                     float* __restrict__ out) {
    __shared__ float  Ksh32[2][BC][KSTRIDE];  // double-buffered K tile (TDM destination)
    __shared__ __bf16 Vsh[2][DH][VPAD];       // double-buffered V tile bf16 transposed
    __shared__ __bf16 Psh[4][16][PPAD];       // per-wave P tile [m][k]

    const int lane = threadIdx.x & 31;
    const int wave = threadIdx.x >> 5;
    const int n    = lane & 15;            // column / row-within-half index
    const int hh   = lane >> 4;            // half select (0/1)
    const int h    = blockIdx.y;
    const int q0   = blockIdx.x * BM + wave * 16;

    const float* Qh = q + (size_t)h * SEQ * DH;
    const float* Kh = k + (size_t)h * SEQ * DH;
    const float* Vh = v + (size_t)h * SEQ * DH;
    float*       Oh = out + (size_t)h * SEQ * DH;

    // ---- Q A-fragments in registers: 4 chunks of K=32 (bf16) ----
    // A layout: m = lane&15; elems 0..15 -> K = hh*16 + 0..15
    v16bf qa[4];
    {
        const float* qrow = Qh + (size_t)(q0 + n) * DH;
#pragma unroll
        for (int c = 0; c < 4; ++c) {
            float4 f0 = *(const float4*)(qrow + c * 32 + hh * 16);
            float4 f1 = *(const float4*)(qrow + c * 32 + hh * 16 + 4);
            float4 f2 = *(const float4*)(qrow + c * 32 + hh * 16 + 8);
            float4 f3 = *(const float4*)(qrow + c * 32 + hh * 16 + 12);
            qa[c][0]  = (__bf16)f0.x; qa[c][1]  = (__bf16)f0.y;
            qa[c][2]  = (__bf16)f0.z; qa[c][3]  = (__bf16)f0.w;
            qa[c][4]  = (__bf16)f1.x; qa[c][5]  = (__bf16)f1.y;
            qa[c][6]  = (__bf16)f1.z; qa[c][7]  = (__bf16)f1.w;
            qa[c][8]  = (__bf16)f2.x; qa[c][9]  = (__bf16)f2.y;
            qa[c][10] = (__bf16)f2.z; qa[c][11] = (__bf16)f2.w;
            qa[c][12] = (__bf16)f3.x; qa[c][13] = (__bf16)f3.y;
            qa[c][14] = (__bf16)f3.z; qa[c][15] = (__bf16)f3.w;
        }
    }

    // ---- flash-attention state ----
    v8f oacc[8] = {};                 // O tile: 16 x 128 f32 (8 d-tiles of v8f)
    float mrow[8], lrow[8];
#pragma unroll
    for (int r = 0; r < 8; ++r) { mrow[r] = -__builtin_inff(); lrow[r] = 0.0f; }

    const int ldkey  = threadIdx.x >> 2;        // 0..31 : key row this thread loads
    const int lddoff = (threadIdx.x & 3) * 32;  // d-range [lddoff, lddoff+32)

    // ---- tile loaders ----
    auto load_k_tile = [&](int kb, int buf) {
#if USE_TDM
        if (wave == 0)
            tdm_load_k_tile(Kh + (size_t)(kb * BC) * DH,
                            (unsigned)(size_t)&Ksh32[buf][0][0]);
#else
        const float* kp = Kh + (size_t)(kb * BC + ldkey) * DH + lddoff;
#pragma unroll
        for (int i = 0; i < 8; ++i)
            *(float4*)&Ksh32[buf][ldkey][lddoff + i * 4] = *(const float4*)(kp + i * 4);
#endif
    };
    auto load_v_tile = [&](int kb, int buf) {
        const float* vp = Vh + (size_t)(kb * BC + ldkey) * DH + lddoff;
#pragma unroll
        for (int i = 0; i < 8; ++i) {
            float4 vf = *(const float4*)(vp + i * 4);
            Vsh[buf][lddoff + i * 4 + 0][ldkey] = (__bf16)vf.x;
            Vsh[buf][lddoff + i * 4 + 1][ldkey] = (__bf16)vf.y;
            Vsh[buf][lddoff + i * 4 + 2][ldkey] = (__bf16)vf.z;
            Vsh[buf][lddoff + i * 4 + 3][ldkey] = (__bf16)vf.w;
        }
    };

    // ---- prologue: fill buffer 0 ----
    load_k_tile(0, 0);
    load_v_tile(0, 0);
#if USE_TDM && __has_builtin(__builtin_amdgcn_s_wait_tensorcnt)
    if (wave == 0)
        __builtin_amdgcn_s_wait_tensorcnt(0);
#endif
    __syncthreads();

    for (int kb = 0; kb < NKB; ++kb) {
        const int cur = kb & 1;
        const int nxt = cur ^ 1;

        // ---- prefetch next tile into the other buffer; DMA hides behind compute ----
        if (kb + 1 < NKB) {
            load_k_tile(kb + 1, nxt);
            load_v_tile(kb + 1, nxt);
        }

        // ---- S = Q*K^T * scale + mask, two 16x16 n-tiles ----
        float sv[2][8];
#pragma unroll
        for (int t = 0; t < 2; ++t) {
            // B layout: n = lane&15 (key col), K base = hh*16, 16 consecutive K per chunk
            const float* krow = &Ksh32[cur][t * 16 + n][hh * 16];
            v16bf kf[4];
#pragma unroll
            for (int c = 0; c < 4; ++c) {
                float4 f0 = *(const float4*)(krow + c * 32);
                float4 f1 = *(const float4*)(krow + c * 32 + 4);
                float4 f2 = *(const float4*)(krow + c * 32 + 8);
                float4 f3 = *(const float4*)(krow + c * 32 + 12);
                kf[c][0]  = (__bf16)f0.x; kf[c][1]  = (__bf16)f0.y;
                kf[c][2]  = (__bf16)f0.z; kf[c][3]  = (__bf16)f0.w;
                kf[c][4]  = (__bf16)f1.x; kf[c][5]  = (__bf16)f1.y;
                kf[c][6]  = (__bf16)f1.z; kf[c][7]  = (__bf16)f1.w;
                kf[c][8]  = (__bf16)f2.x; kf[c][9]  = (__bf16)f2.y;
                kf[c][10] = (__bf16)f2.z; kf[c][11] = (__bf16)f2.w;
                kf[c][12] = (__bf16)f3.x; kf[c][13] = (__bf16)f3.y;
                kf[c][14] = (__bf16)f3.z; kf[c][15] = (__bf16)f3.w;
            }
            v8f sacc = {};
#pragma unroll
            for (int c = 0; c < 4; ++c)
                sacc = __builtin_amdgcn_wmma_f32_16x16x32_bf16(
                    false, qa[c], false, kf[c], (short)0, sacc, false, false);
            const int kcol = kb * BC + t * 16 + n;
            const float* mp = mask + (size_t)(q0 + 8 * hh) * SEQ + kcol;
#pragma unroll
            for (int r = 0; r < 8; ++r)
                sv[t][r] = sacc[r] * SCALE + mp[(size_t)r * SEQ];
        }

        // ---- online softmax: row max over the 16 lanes of each half ----
        float mnew[8], corr[8];
#pragma unroll
        for (int r = 0; r < 8; ++r) {
            float mx = fmaxf(sv[0][r], sv[1][r]);
#pragma unroll
            for (int i = 1; i < 16; i <<= 1) mx = fmaxf(mx, __shfl_xor(mx, i, 32));
            mnew[r] = fmaxf(mrow[r], mx);
            corr[r] = __expf(mrow[r] - mnew[r]);
            mrow[r] = mnew[r];
        }

        // ---- p = exp(s - m); l from pre-dropout p; dropped/scaled p -> LDS bf16 ----
        float ps[2][8];
#pragma unroll
        for (int t = 0; t < 2; ++t) {
            const int kcol = kb * BC + t * 16 + n;
#pragma unroll
            for (int r = 0; r < 8; ++r) {
                float p = __expf(sv[t][r] - mnew[r]);
                ps[t][r] = p;
                unsigned int idx = (unsigned int)(h * SEQ * SEQ)
                                 + (unsigned int)((q0 + r + 8 * hh) * SEQ) + (unsigned int)kcol;
                unsigned int x = mix_hash(idx);
                float keep = ((x >> 8) < KEEP_THRESH) ? INV_KEEP : 0.0f;
                Psh[wave][r + 8 * hh][t * 16 + n] = (__bf16)(p * keep);
            }
        }
#pragma unroll
        for (int r = 0; r < 8; ++r) {
            float s = ps[0][r] + ps[1][r];
#pragma unroll
            for (int i = 1; i < 16; i <<= 1) s += __shfl_xor(s, i, 32);
            lrow[r] = lrow[r] * corr[r] + s;
        }
#pragma unroll
        for (int dt = 0; dt < 8; ++dt)
#pragma unroll
            for (int r = 0; r < 8; ++r)
                oacc[dt][r] *= corr[r];

        // ---- reload P in A-fragment layout (per-wave tile, in-order LDS) ----
        v16bf pf;
        {
            const __bf16* prow = &Psh[wave][n][hh * 16];
            *(uint4*)&pf       = *(const uint4*)(prow);
            *((uint4*)&pf + 1) = *(const uint4*)(prow + 8);
        }

        // ---- O += P * V : software-pipelined V fragment loads + 8 WMMAs ----
        v16bf vf = *(const v16bf*)&Vsh[cur][n][hh * 16];
#pragma unroll
        for (int dt = 0; dt < 8; ++dt) {
            v16bf curf = vf;
            if (dt < 7)
                vf = *(const v16bf*)&Vsh[cur][(dt + 1) * 16 + n][hh * 16];
            oacc[dt] = __builtin_amdgcn_wmma_f32_16x16x32_bf16(
                false, pf, false, curf, (short)0, oacc[dt], false, false);
        }

        // ---- publish next buffer: DMA completion + LDS fence + barrier ----
#if USE_TDM && __has_builtin(__builtin_amdgcn_s_wait_tensorcnt)
        if (wave == 0)
            __builtin_amdgcn_s_wait_tensorcnt(0);
#endif
        __syncthreads();
    }

    // ---- epilogue: normalize by l and store fp32 ----
#pragma unroll
    for (int r = 0; r < 8; ++r) {
        float inv = 1.0f / lrow[r];
        float* orow = Oh + (size_t)(q0 + r + 8 * hh) * DH + n;
#pragma unroll
        for (int dt = 0; dt < 8; ++dt)
            orow[dt * 16] = oacc[dt][r] * inv;
    }
}

extern "C" void kernel_launch(void* const* d_in, const int* in_sizes, int n_in,
                              void* d_out, int out_size, void* d_ws, size_t ws_size,
                              hipStream_t stream) {
    const float* q    = (const float*)d_in[0];
    const float* k    = (const float*)d_in[1];
    const float* v    = (const float*)d_in[2];
    const float* mask = (const float*)d_in[3];
    float* out = (float*)d_out;
    dim3 grid(SEQ / BM, NHEAD, 1);
    dim3 block(128, 1, 1);
    fattn_bf16_wmma<<<grid, block, 0, stream>>>(q, k, v, mask, out);
}
